// RNAStructureModule_83691732730287
// MI455X (gfx1250) — matrix operand order repaired and checked by reading, not compiled, not run on hardware
//
#include <hip/hip_runtime.h>
#include <hip/hip_bf16.h>
#include <cstdint>

typedef __attribute__((ext_vector_type(16))) _Float16 v16h;
typedef __attribute__((ext_vector_type(8)))  _Float16 v8h;
typedef __attribute__((ext_vector_type(8)))  float    v8f;

constexpr int NT  = 768;     // tokens
constexpr int CS  = 384;
constexpr int CZ  = 128;
constexpr int H   = 12;
constexpr int C   = 16;
constexpr int PQK = 4;
constexpr int PV  = 8;
constexpr int L   = 8;
constexpr int CAT = H*C + H*PV*3 + H*PV + H*CZ;  // 2112

constexpr float WL = 0.5773502691896258f;        // 1/sqrt(3)
constexpr float WC = 0.23570226039551584f;       // sqrt(2/(9*PQK))

// ---------------------------------------------------------------------------
// Fragment loaders (wave32, v_wmma_f32_16x16x32_f16 operand layouts)
// A (16x32 f16): lane = row m0+(lane&15); halves 0..7 -> K = kb + 8*g + t,
//                halves 8..15 -> K = kb + 16 + 8*g + t   (g = lane>=16)
// B (32x16 f16): lane = col n0+(lane&15); same K mapping as A
// C/D (16x16 f32): acc[j] -> row m0 + j + 8*g, col n0 + (lane&15)
// ---------------------------------------------------------------------------
__device__ inline v16h frag_rowmajor(const _Float16* __restrict__ A, int ld,
                                     int m0, int kb, int lane) {
  int r = m0 + (lane & 15);
  const _Float16* p = A + (size_t)r * ld + kb + ((lane >> 4) << 3);
  v8h lo = *(const v8h*)p;
  v8h hi = *(const v8h*)(p + 16);
  v16h out;
#pragma unroll
  for (int i = 0; i < 8; ++i) { out[i] = lo[i]; out[i + 8] = hi[i]; }
  return out;
}

__device__ inline v16h fragB(const _Float16* __restrict__ B, int ld, int ncols,
                             int n0, int kb, int lane) {
  int c = n0 + (lane & 15);
  v16h out = {};
  if (c < ncols) {
    const _Float16* p = B + (size_t)(kb + ((lane >> 4) << 3)) * ld + c;
#pragma unroll
    for (int i = 0; i < 8; ++i) { out[i] = p[(size_t)i * ld]; out[i + 8] = p[(size_t)(i + 16) * ld]; }
  }
  return out;
}

__device__ inline v8f wmma32(v16h a, v16h b, v8f c) {
  return __builtin_amdgcn_wmma_f32_16x16x32_f16(false, a, false, b, (short)0, c,
                                                false, false);
}

// gfx1250 async DMA: global -> LDS, tracked by ASYNCcnt
__device__ inline void async_g2lds_b32(unsigned lds_byte_off,
                                       const float* gaddr) {
  asm volatile("global_load_async_to_lds_b32 %0, %1, off"
               :: "v"(lds_byte_off), "v"(gaddr) : "memory");
}
__device__ inline void wait_asynccnt0() {
  asm volatile("s_wait_asynccnt 0x0" ::: "memory");
}

// ---------------------------------------------------------------------------
// Generic wave-per-tile GEMM: C[M,ncols] = A[M,K] (f16,row-major) * B[K,ncols]
// ---------------------------------------------------------------------------
__global__ void gemm16_kernel(const _Float16* __restrict__ A, int lda,
                              const _Float16* __restrict__ B, int ldb, int ncols,
                              int K, int ntn,
                              float* __restrict__ out32,
                              _Float16* __restrict__ out16,
                              int ldo, float scale, int nwaves) {
  int wid  = (int)((blockIdx.x * blockDim.x + threadIdx.x) >> 5);
  int lane = threadIdx.x & 31;
  if (wid >= nwaves) return;
  int nt = wid % ntn, mt = wid / ntn;
  int m0 = mt * 16, n0 = nt * 16;
  v8f acc = {};
  for (int kb = 0; kb < K; kb += 32) {
    if (kb + 32 < K)
      __builtin_prefetch(A + (size_t)(m0 + (lane & 15)) * lda + kb + 32, 0, 1);
    v16h a = frag_rowmajor(A, lda, m0, kb, lane);
    v16h b = fragB(B, ldb, ncols, n0, kb, lane);
    acc = wmma32(a, b, acc);
  }
  int c = n0 + (lane & 15);
  int g = lane >> 4;
  if (c < ncols) {
#pragma unroll
    for (int j = 0; j < 8; ++j) {
      int r = m0 + j + 8 * g;
      float v = acc[j] * scale;
      if (out32) out32[(size_t)r * ldo + c] = v;
      if (out16) out16[(size_t)r * ldo + c] = (_Float16)v;
    }
  }
}

// ---------------------------------------------------------------------------
// Conversions
// ---------------------------------------------------------------------------
__global__ void cvt_f32_f16(const float* __restrict__ src,
                            _Float16* __restrict__ dst, size_t n) {
  size_t i = (size_t)blockIdx.x * blockDim.x + threadIdx.x;
  if (i < n) dst[i] = (_Float16)src[i];
}

__global__ void cvt_pad_f16(const float* __restrict__ src,
                            _Float16* __restrict__ dst,
                            int rows, int cols, int padcols) {
  int i = blockIdx.x * blockDim.x + threadIdx.x;
  if (i >= rows * padcols) return;
  int r = i / padcols, c = i % padcols;
  dst[i] = (c < cols) ? (_Float16)src[(size_t)r * cols + c] : (_Float16)0.f;
}

__global__ void init_s_kernel(const float* __restrict__ s_in,
                              float* __restrict__ s_out,
                              _Float16* __restrict__ s16, int n) {
  int i = blockIdx.x * blockDim.x + threadIdx.x;
  if (i >= n) return;
  float v = s_in[i];
  s_out[i] = v;
  s16[i] = (_Float16)v;
}

// ---------------------------------------------------------------------------
// Point transform: local->global frames, squared norms, padded f16 vpg
// ---------------------------------------------------------------------------
__global__ void point_kernel(const float* __restrict__ rot,
                             const float* __restrict__ trans,
                             const float* __restrict__ qp,
                             const float* __restrict__ kp,
                             const float* __restrict__ vp,
                             float* __restrict__ qpg,
                             _Float16* __restrict__ kpg16,
                             _Float16* __restrict__ vpg16,
                             float* __restrict__ qn, float* __restrict__ kn) {
  int t = blockIdx.x * blockDim.x + threadIdx.x;
  if (t >= NT * H) return;
  int n = t / H, h = t % H;
  float R[9], tr[3];
#pragma unroll
  for (int i = 0; i < 9; ++i) R[i] = rot[(size_t)n * 9 + i];
#pragma unroll
  for (int i = 0; i < 3; ++i) tr[i] = trans[(size_t)n * 3 + i];

  float qs = 0.f, ks = 0.f;
#pragma unroll
  for (int p = 0; p < PQK; ++p) {
    const float* lp = qp + (size_t)n * (H * PQK * 3) + (h * PQK + p) * 3;
    float x = lp[0], y = lp[1], z0 = lp[2];
    float gx = R[0]*x + R[1]*y + R[2]*z0 + tr[0];
    float gy = R[3]*x + R[4]*y + R[5]*z0 + tr[1];
    float gz = R[6]*x + R[7]*y + R[8]*z0 + tr[2];
    float* dst = qpg + ((size_t)n * H + h) * 12 + p * 3;
    dst[0] = gx; dst[1] = gy; dst[2] = gz;
    qs += gx*gx + gy*gy + gz*gz;
  }
#pragma unroll
  for (int p = 0; p < PQK; ++p) {
    const float* lp = kp + (size_t)n * (H * PQK * 3) + (h * PQK + p) * 3;
    float x = lp[0], y = lp[1], z0 = lp[2];
    float gx = R[0]*x + R[1]*y + R[2]*z0 + tr[0];
    float gy = R[3]*x + R[4]*y + R[5]*z0 + tr[1];
    float gz = R[6]*x + R[7]*y + R[8]*z0 + tr[2];
    _Float16* dst = kpg16 + ((size_t)n * H + h) * 12 + p * 3;
    dst[0] = (_Float16)gx; dst[1] = (_Float16)gy; dst[2] = (_Float16)gz;
    ks += gx*gx + gy*gy + gz*gz;
  }
#pragma unroll
  for (int p = 0; p < PV; ++p) {
    const float* lp = vp + (size_t)n * (H * PV * 3) + (h * PV + p) * 3;
    float x = lp[0], y = lp[1], z0 = lp[2];
    float gx = R[0]*x + R[1]*y + R[2]*z0 + tr[0];
    float gy = R[3]*x + R[4]*y + R[5]*z0 + tr[1];
    float gz = R[6]*x + R[7]*y + R[8]*z0 + tr[2];
    _Float16* dst = vpg16 + (size_t)n * (H * 32) + h * 32 + p * 3;
    dst[0] = (_Float16)gx; dst[1] = (_Float16)gy; dst[2] = (_Float16)gz;
  }
#pragma unroll
  for (int u = 24; u < 32; ++u)
    vpg16[(size_t)n * (H * 32) + h * 32 + u] = (_Float16)0.f;
  qn[t] = qs;
  kn[t] = ks;
}

// ---------------------------------------------------------------------------
// bias[i,j,h] = sum_c z16[i,j,c] * wb16[c,h]   (wave per (i, j-tile))
// ---------------------------------------------------------------------------
__global__ void bias_kernel(const _Float16* __restrict__ z16,
                            const _Float16* __restrict__ wb16,
                            float* __restrict__ bias, int nwaves) {
  int wid  = (int)((blockIdx.x * blockDim.x + threadIdx.x) >> 5);
  int lane = threadIdx.x & 31;
  if (wid >= nwaves) return;
  int jt = wid % (NT / 16);
  int i  = wid / (NT / 16);
  const _Float16* A = z16 + (size_t)i * NT * CZ;
  v8f acc = {};
#pragma unroll
  for (int kb = 0; kb < CZ; kb += 32) {
    v16h a = frag_rowmajor(A, CZ, jt * 16, kb, lane);
    v16h b = fragB(wb16, 16, 16, 0, kb, lane);
    acc = wmma32(a, b, acc);
  }
  int hh = lane & 15;
  int g  = lane >> 4;
  if (hh < H) {
#pragma unroll
    for (int j = 0; j < 8; ++j) {
      int jr = jt * 16 + j + 8 * g;
      bias[((size_t)i * NT + jr) * H + hh] = acc[j];
    }
  }
}

// ---------------------------------------------------------------------------
// Fused score: one K=32 WMMA = q.k (16 dims, prescaled wL/sqrtC) + point-cross
// (12 dims, A scaled by wL*gamma*wC).  Epilogue adds wL*bias - coef*(qn+kn).
// ---------------------------------------------------------------------------
__global__ void score_kernel(const _Float16* __restrict__ q16,
                             const _Float16* __restrict__ k16,
                             const float* __restrict__ qpg,
                             const _Float16* __restrict__ kpg16,
                             const float* __restrict__ qn,
                             const float* __restrict__ kn,
                             const float* __restrict__ bias,
                             const float* __restrict__ head_w_l,
                             float* __restrict__ att, int nwaves) {
  int wid  = (int)((blockIdx.x * blockDim.x + threadIdx.x) >> 5);
  int lane = threadIdx.x & 31;
  if (wid >= nwaves) return;
  int jt = wid % (NT / 16);
  int it = (wid / (NT / 16)) % (NT / 16);
  int h  = wid / ((NT / 16) * (NT / 16));
  int g  = lane >> 4;

  float hw    = head_w_l[h];
  float gamma = (hw > 20.f) ? hw : log1pf(__expf(hw));
  float csc   = WL * gamma * WC;       // cross coefficient
  float halfd = 0.5f * csc;            // coefficient on qn+kn

  int mrow = it * 16 + (lane & 15);
  v16h a;
  {
    const _Float16* pq = q16 + (size_t)mrow * (H * C) + h * C + g * 8;
    v8h lo = *(const v8h*)pq;
#pragma unroll
    for (int t = 0; t < 8; ++t) a[t] = lo[t];
    const float* pp = qpg + ((size_t)mrow * H + h) * 12;
#pragma unroll
    for (int t = 0; t < 8; ++t) {
      int pd = g * 8 + t;
      a[8 + t] = (pd < 12) ? (_Float16)(pp[pd] * csc) : (_Float16)0.f;
    }
  }
  int ncol = jt * 16 + (lane & 15);
  v16h b;
  {
    const _Float16* pk = k16 + (size_t)ncol * (H * C) + h * C + g * 8;
    v8h lo = *(const v8h*)pk;
    const _Float16* pp = kpg16 + ((size_t)ncol * H + h) * 12;
#pragma unroll
    for (int t = 0; t < 8; ++t) {
      int pd = g * 8 + t;
      b[t] = lo[t];
      b[8 + t] = (pd < 12) ? pp[pd] : (_Float16)0.f;
    }
  }
  v8f acc = {};
  acc = wmma32(a, b, acc);

  float knj = kn[(size_t)ncol * H + h];
#pragma unroll
  for (int j = 0; j < 8; ++j) {
    int ir = it * 16 + j + 8 * g;
    float lg = acc[j] + WL * bias[((size_t)ir * NT + ncol) * H + h]
             - halfd * (qn[(size_t)ir * H + h] + knj);
    att[((size_t)h * NT + ir) * NT + ncol] = lg;
  }
}

// ---------------------------------------------------------------------------
// Row softmax over N=768.  The row is staged into LDS with gfx1250 async
// global->LDS DMA (ASYNCcnt), then both passes read LDS.
// ---------------------------------------------------------------------------
__global__ void softmax_kernel(const float* __restrict__ att,
                               _Float16* __restrict__ att16) {
  int row = blockIdx.x;                 // h*NT + i
  int tid = threadIdx.x;
  const float* p = att + (size_t)row * NT;
  __shared__ float sbuf[NT];
  __shared__ float red[256];

  for (int j = tid; j < NT; j += 256)
    async_g2lds_b32((unsigned)(uintptr_t)(&sbuf[j]), p + j);
  wait_asynccnt0();
  __syncthreads();

  float m = -1e30f;
  for (int j = tid; j < NT; j += 256) m = fmaxf(m, sbuf[j]);
  red[tid] = m;
  __syncthreads();
  for (int s = 128; s > 0; s >>= 1) {
    if (tid < s) red[tid] = fmaxf(red[tid], red[tid + s]);
    __syncthreads();
  }
  float mx = red[0];
  __syncthreads();
  float sum = 0.f;
  for (int j = tid; j < NT; j += 256) {
    float v = __expf(sbuf[j] - mx);
    sbuf[j] = v;                        // each thread touches only its own j's
    sum += v;
  }
  red[tid] = sum;
  __syncthreads();
  for (int s = 128; s > 0; s >>= 1) {
    if (tid < s) red[tid] += red[tid + s];
    __syncthreads();
  }
  float inv = 1.f / red[0];
  for (int j = tid; j < NT; j += 256)
    att16[(size_t)row * NT + j] = (_Float16)(sbuf[j] * inv);
}

// ---------------------------------------------------------------------------
// o = a.v  and  optg = a.vpg   (wave per (h, i-tile, ntile))
// ---------------------------------------------------------------------------
__global__ void o_optg_kernel(const _Float16* __restrict__ att16,
                              const _Float16* __restrict__ v16,
                              const _Float16* __restrict__ vpg16,
                              float* __restrict__ o, float* __restrict__ optg,
                              int nwaves) {
  int wid  = (int)((blockIdx.x * blockDim.x + threadIdx.x) >> 5);
  int lane = threadIdx.x & 31;
  if (wid >= nwaves) return;
  int nt = wid % 3;
  int it = (wid / 3) % (NT / 16);
  int h  = wid / (3 * (NT / 16));
  int g  = lane >> 4;

  const _Float16* A = att16 + (size_t)h * NT * NT;
  const _Float16* B;
  int ldb;
  if (nt == 0) { B = v16 + h * C;                 ldb = H * C;  }
  else         { B = vpg16 + h * 32 + (nt - 1) * 16; ldb = H * 32; }

  v8f acc = {};
  for (int kb = 0; kb < NT; kb += 32) {
    v16h a = frag_rowmajor(A, NT, it * 16, kb, lane);
    v16h b = fragB(B, ldb, 16, 0, kb, lane);
    acc = wmma32(a, b, acc);
  }
  int cc = lane & 15;
#pragma unroll
  for (int j = 0; j < 8; ++j) {
    int r = it * 16 + j + 8 * g;
    if (nt == 0) {
      o[(size_t)r * (H * C) + h * C + cc] = acc[j];
    } else {
      int pc = (nt - 1) * 16 + cc;
      if (pc < 24) optg[(size_t)r * (H * 24) + h * 24 + pc] = acc[j];
    }
  }
}

// ---------------------------------------------------------------------------
// opair[i,h,c] = sum_j a[h,i,j] z16[i,j,c]:  M = heads (pad 16), K = j, N = c
// ---------------------------------------------------------------------------
__global__ void opair_kernel(const _Float16* __restrict__ att16,
                             const _Float16* __restrict__ z16,
                             float* __restrict__ opair, int nwaves) {
  int wid  = (int)((blockIdx.x * blockDim.x + threadIdx.x) >> 5);
  int lane = threadIdx.x & 31;
  if (wid >= nwaves) return;
  int nt = wid % (CZ / 16);
  int i  = wid / (CZ / 16);
  int g  = lane >> 4;
  int m  = lane & 15;

  const _Float16* Bz = z16 + (size_t)i * NT * CZ;
  v8f acc = {};
  for (int kb = 0; kb < NT; kb += 32) {
    v16h a = {};
    if (m < H) {
      const _Float16* p = att16 + ((size_t)m * NT + i) * NT + kb + g * 8;
      v8h lo = *(const v8h*)p;
      v8h hi = *(const v8h*)(p + 16);
#pragma unroll
      for (int t = 0; t < 8; ++t) { a[t] = lo[t]; a[t + 8] = hi[t]; }
    }
    v16h b = fragB(Bz, CZ, CZ, nt * 16, kb, lane);
    acc = wmma32(a, b, acc);
  }
  int c = nt * 16 + (lane & 15);
#pragma unroll
  for (int j = 0; j < 8; ++j) {
    int mh = j + 8 * g;
    if (mh < H) opair[((size_t)i * H + mh) * CZ + c] = acc[j];
  }
}

// ---------------------------------------------------------------------------
// Build concatenated f16 feature: [o | R^T(optg-t) | norm | opair]
// ---------------------------------------------------------------------------
__global__ void cat_kernel(const float* __restrict__ o,
                           const float* __restrict__ optg,
                           const float* __restrict__ opair,
                           const float* __restrict__ rot,
                           const float* __restrict__ trans,
                           _Float16* __restrict__ cat16) {
  int idx = blockIdx.x * blockDim.x + threadIdx.x;
  if (idx >= NT * CAT) return;
  int n = idx / CAT, col = idx % CAT;
  float val;
  if (col < H * C) {
    val = o[(size_t)n * (H * C) + col];
  } else if (col < H * C + H * PV * 3) {
    int u = col - H * C;
    int hh = u / 24, rem = u % 24, p = rem / 3, x = rem % 3;
    const float* gp = optg + (size_t)n * (H * 24) + hh * 24 + p * 3;
    float w0 = gp[0] - trans[(size_t)n * 3 + 0];
    float w1 = gp[1] - trans[(size_t)n * 3 + 1];
    float w2 = gp[2] - trans[(size_t)n * 3 + 2];
    val = rot[(size_t)n * 9 + 0 * 3 + x] * w0 +
          rot[(size_t)n * 9 + 1 * 3 + x] * w1 +
          rot[(size_t)n * 9 + 2 * 3 + x] * w2;   // R^T (x - t)
  } else if (col < H * C + H * PV * 3 + H * PV) {
    int u = col - (H * C + H * PV * 3);
    int hh = u / PV, p = u % PV;
    const float* gp = optg + (size_t)n * (H * 24) + hh * 24 + p * 3;
    float w0 = gp[0] - trans[(size_t)n * 3 + 0];
    float w1 = gp[1] - trans[(size_t)n * 3 + 1];
    float w2 = gp[2] - trans[(size_t)n * 3 + 2];
    val = sqrtf(w0 * w0 + w1 * w1 + w2 * w2 + 1e-8f);  // rotation-invariant
  } else {
    val = opair[(size_t)n * (H * CZ) + (col - (H * C + H * PV * 3 + H * PV))];
  }
  cat16[(size_t)n * CAT + col] = (_Float16)val;
}

// ---------------------------------------------------------------------------
// Output projection + residual: s += cat @ wout + bout ; refresh s16
// ---------------------------------------------------------------------------
__global__ void out_kernel(const _Float16* __restrict__ cat16,
                           const _Float16* __restrict__ wout16,
                           const float* __restrict__ bout,
                           float* __restrict__ s, _Float16* __restrict__ s16,
                           int nwaves) {
  int wid  = (int)((blockIdx.x * blockDim.x + threadIdx.x) >> 5);
  int lane = threadIdx.x & 31;
  if (wid >= nwaves) return;
  int nt = wid % (CS / 16);
  int mt = wid / (CS / 16);
  int m0 = mt * 16, n0 = nt * 16;
  int g = lane >> 4;
  v8f acc = {};
  for (int kb = 0; kb < CAT; kb += 32) {
    if (kb + 32 < CAT)
      __builtin_prefetch(cat16 + (size_t)(m0 + (lane & 15)) * CAT + kb + 32, 0, 1);
    v16h a = frag_rowmajor(cat16, CAT, m0, kb, lane);
    v16h b = fragB(wout16, CS, CS, n0, kb, lane);
    acc = wmma32(a, b, acc);
  }
  int c = n0 + (lane & 15);
  float bo = bout[c];
#pragma unroll
  for (int j = 0; j < 8; ++j) {
    int r = m0 + j + 8 * g;
    float v = s[(size_t)r * CS + c] + acc[j] + bo;
    s[(size_t)r * CS + c] = v;
    s16[(size_t)r * CS + c] = (_Float16)v;
  }
}

// ---------------------------------------------------------------------------
// Base-pair head (fused): sigmoid(relu(z @ w1 + b1) @ w2 + b2)
// wave per 16 flattened (i,j) rows; shuffle-reduce the 64 hidden dims
// ---------------------------------------------------------------------------
__global__ void bp_kernel(const _Float16* __restrict__ z16,
                          const _Float16* __restrict__ w1_16,
                          const float* __restrict__ b1,
                          const float* __restrict__ w2,
                          const float* __restrict__ b2,
                          float* __restrict__ bp, int nwaves) {
  int wid  = (int)((blockIdx.x * blockDim.x + threadIdx.x) >> 5);
  int lane = threadIdx.x & 31;
  if (wid >= nwaves) return;
  int r0 = wid * 16;
  int g = lane >> 4;
  float rowsum[8];
#pragma unroll
  for (int j = 0; j < 8; ++j) rowsum[j] = 0.f;

#pragma unroll
  for (int t = 0; t < 4; ++t) {          // 4 n-tiles of 16 over 64 hidden
    v8f acc = {};
#pragma unroll
    for (int kb = 0; kb < CZ; kb += 32) {
      v16h a = frag_rowmajor(z16, CZ, r0, kb, lane);
      v16h b = fragB(w1_16, 64, 64, t * 16, kb, lane);
      acc = wmma32(a, b, acc);
    }
    int n = t * 16 + (lane & 15);
    float w2n = w2[n];
    float b1n = b1[n];
#pragma unroll
    for (int j = 0; j < 8; ++j) {
      float h1 = fmaxf(acc[j] + b1n, 0.f);
      rowsum[j] += h1 * w2n;
    }
  }
  float bb = b2[0];
#pragma unroll
  for (int j = 0; j < 8; ++j) {
    float v = rowsum[j];
    v += __shfl_xor(v, 1, 32);
    v += __shfl_xor(v, 2, 32);
    v += __shfl_xor(v, 4, 32);
    v += __shfl_xor(v, 8, 32);
    if ((lane & 15) == 0) {
      int r = r0 + j + 8 * g;
      bp[r] = 1.f / (1.f + __expf(-(v + bb)));
    }
  }
}

// ---------------------------------------------------------------------------
// Torsions: s16 @ tor_w (padded to 16 cols) + tor_b
// ---------------------------------------------------------------------------
__global__ void torsion_kernel(const _Float16* __restrict__ s16,
                               const _Float16* __restrict__ tor16,
                               const float* __restrict__ tor_b,
                               float* __restrict__ tor_out, int nwaves) {
  int wid  = (int)((blockIdx.x * blockDim.x + threadIdx.x) >> 5);
  int lane = threadIdx.x & 31;
  if (wid >= nwaves) return;
  int m0 = wid * 16;
  int g = lane >> 4;
  v8f acc = {};
  for (int kb = 0; kb < CS; kb += 32) {
    v16h a = frag_rowmajor(s16, CS, m0, kb, lane);
    v16h b = fragB(tor16, 16, 16, 0, kb, lane);
    acc = wmma32(a, b, acc);
  }
  int c = lane & 15;
  if (c < 7) {
    float tb = tor_b[c];
#pragma unroll
    for (int j = 0; j < 8; ++j) {
      int r = m0 + j + 8 * g;
      tor_out[(size_t)r * 7 + c] = acc[j] + tb;
    }
  }
}

// ---------------------------------------------------------------------------
// Host orchestration
// ---------------------------------------------------------------------------
static inline dim3 wave_grid(long long nwaves) {
  return dim3((unsigned)((nwaves * 32 + 255) / 256));
}
static inline dim3 elem_grid(long long n) {
  return dim3((unsigned)((n + 255) / 256));
}

extern "C" void kernel_launch(void* const* d_in, const int* in_sizes, int n_in,
                              void* d_out, int out_size, void* d_ws,
                              size_t ws_size, hipStream_t stream) {
  (void)in_sizes; (void)n_in; (void)out_size; (void)ws_size;

  const float* s_in   = (const float*)d_in[0];
  const float* z      = (const float*)d_in[1];
  const float* rot    = (const float*)d_in[2];
  const float* trans  = (const float*)d_in[3];
  const float* wq     = (const float*)d_in[4];
  const float* wk     = (const float*)d_in[5];
  const float* wv     = (const float*)d_in[6];
  const float* wqp    = (const float*)d_in[7];
  const float* wkp    = (const float*)d_in[8];
  const float* wvp    = (const float*)d_in[9];
  const float* wb     = (const float*)d_in[10];
  const float* head_w = (const float*)d_in[11];
  const float* wout   = (const float*)d_in[12];
  const float* bout   = (const float*)d_in[13];
  const float* bp_w1  = (const float*)d_in[14];
  const float* bp_b1  = (const float*)d_in[15];
  const float* bp_w2  = (const float*)d_in[16];
  const float* bp_b2  = (const float*)d_in[17];
  const float* tor_w  = (const float*)d_in[18];
  const float* tor_b  = (const float*)d_in[19];

  float* s_out   = (float*)d_out;                   // [768,384]
  float* bp_out  = s_out + (size_t)NT * CS;         // [768,768]
  float* tor_out = bp_out + (size_t)NT * NT;        // [768,7]

  // ---- workspace carve-out (256B aligned) ----
  char* ws = (char*)d_ws;
  size_t off = 0;
  auto alloc = [&](size_t bytes) -> void* {
    off = (off + 255) & ~(size_t)255;
    void* p = ws + off;
    off += bytes;
    return p;
  };
  _Float16* z16    = (_Float16*)alloc((size_t)NT * NT * CZ * 2);
  _Float16* wq16   = (_Float16*)alloc((size_t)L * CS * H * C * 2);
  _Float16* wk16   = (_Float16*)alloc((size_t)L * CS * H * C * 2);
  _Float16* wv16   = (_Float16*)alloc((size_t)L * CS * H * C * 2);
  _Float16* wqp16  = (_Float16*)alloc((size_t)L * CS * H * PQK * 3 * 2);
  _Float16* wkp16  = (_Float16*)alloc((size_t)L * CS * H * PQK * 3 * 2);
  _Float16* wvp16  = (_Float16*)alloc((size_t)L * CS * H * PV * 3 * 2);
  _Float16* wb16   = (_Float16*)alloc((size_t)L * CZ * 16 * 2);
  _Float16* wout16 = (_Float16*)alloc((size_t)L * CAT * CS * 2);
  _Float16* w1_16  = (_Float16*)alloc((size_t)CZ * 64 * 2);
  _Float16* tor16  = (_Float16*)alloc((size_t)CS * 16 * 2);
  _Float16* s16    = (_Float16*)alloc((size_t)NT * CS * 2);
  _Float16* q16    = (_Float16*)alloc((size_t)NT * H * C * 2);
  _Float16* k16    = (_Float16*)alloc((size_t)NT * H * C * 2);
  _Float16* v16    = (_Float16*)alloc((size_t)NT * H * C * 2);
  float*    qp     = (float*)alloc((size_t)NT * H * PQK * 3 * 4);
  float*    kp     = (float*)alloc((size_t)NT * H * PQK * 3 * 4);
  float*    vp     = (float*)alloc((size_t)NT * H * PV * 3 * 4);
  float*    qpg    = (float*)alloc((size_t)NT * H * 12 * 4);
  _Float16* kpg16  = (_Float16*)alloc((size_t)NT * H * 12 * 2);
  _Float16* vpg16  = (_Float16*)alloc((size_t)NT * H * 32 * 2);
  float*    qn     = (float*)alloc((size_t)NT * H * 4);
  float*    kn     = (float*)alloc((size_t)NT * H * 4);
  float*    biasb  = (float*)alloc((size_t)NT * NT * H * 4);
  float*    att    = (float*)alloc((size_t)H * NT * NT * 4);
  _Float16* att16  = (_Float16*)alloc((size_t)H * NT * NT * 2);
  float*    obuf   = (float*)alloc((size_t)NT * H * C * 4);
  float*    optg   = (float*)alloc((size_t)NT * H * 24 * 4);
  float*    opair  = (float*)alloc((size_t)NT * H * CZ * 4);
  _Float16* cat16  = (_Float16*)alloc((size_t)NT * CAT * 2);

  dim3 blk(256);

  // ---- one-time conversions ----
  init_s_kernel<<<elem_grid(NT * CS), blk, 0, stream>>>(s_in, s_out, s16, NT * CS);
  cvt_f32_f16<<<elem_grid((size_t)NT * NT * CZ), blk, 0, stream>>>(z, z16, (size_t)NT * NT * CZ);
  cvt_f32_f16<<<elem_grid((size_t)L * CS * 192), blk, 0, stream>>>(wq, wq16, (size_t)L * CS * 192);
  cvt_f32_f16<<<elem_grid((size_t)L * CS * 192), blk, 0, stream>>>(wk, wk16, (size_t)L * CS * 192);
  cvt_f32_f16<<<elem_grid((size_t)L * CS * 192), blk, 0, stream>>>(wv, wv16, (size_t)L * CS * 192);
  cvt_f32_f16<<<elem_grid((size_t)L * CS * 144), blk, 0, stream>>>(wqp, wqp16, (size_t)L * CS * 144);
  cvt_f32_f16<<<elem_grid((size_t)L * CS * 144), blk, 0, stream>>>(wkp, wkp16, (size_t)L * CS * 144);
  cvt_f32_f16<<<elem_grid((size_t)L * CS * 288), blk, 0, stream>>>(wvp, wvp16, (size_t)L * CS * 288);
  cvt_f32_f16<<<elem_grid((size_t)L * CAT * CS), blk, 0, stream>>>(wout, wout16, (size_t)L * CAT * CS);
  cvt_f32_f16<<<elem_grid((size_t)CZ * 64), blk, 0, stream>>>(bp_w1, w1_16, (size_t)CZ * 64);
  cvt_pad_f16<<<elem_grid((size_t)L * CZ * 16), blk, 0, stream>>>(wb, wb16, L * CZ, H, 16);
  cvt_pad_f16<<<elem_grid((size_t)CS * 16), blk, 0, stream>>>(tor_w, tor16, CS, 7, 16);

  const int MT = NT / 16;  // 48 row tiles

  for (int l = 0; l < L; ++l) {
    const _Float16* wq_l  = wq16  + (size_t)l * CS * 192;
    const _Float16* wk_l  = wk16  + (size_t)l * CS * 192;
    const _Float16* wv_l  = wv16  + (size_t)l * CS * 192;
    const _Float16* wqp_l = wqp16 + (size_t)l * CS * 144;
    const _Float16* wkp_l = wkp16 + (size_t)l * CS * 144;
    const _Float16* wvp_l = wvp16 + (size_t)l * CS * 288;
    const _Float16* wb_l  = wb16  + (size_t)l * CZ * 16;
    const _Float16* wo_l  = wout16 + (size_t)l * CAT * CS;
    const float*    hw_l  = head_w + (size_t)l * H;
    const float*    bo_l  = bout   + (size_t)l * CS;

    // projections (q prescaled by wL/sqrt(C))
    int nw;
    nw = MT * 12;
    gemm16_kernel<<<wave_grid(nw), blk, 0, stream>>>(s16, CS, wq_l, 192, 192, CS, 12,
        nullptr, q16, 192, WL * 0.25f, nw);
    gemm16_kernel<<<wave_grid(nw), blk, 0, stream>>>(s16, CS, wk_l, 192, 192, CS, 12,
        nullptr, k16, 192, 1.f, nw);
    gemm16_kernel<<<wave_grid(nw), blk, 0, stream>>>(s16, CS, wv_l, 192, 192, CS, 12,
        nullptr, v16, 192, 1.f, nw);
    nw = MT * 9;
    gemm16_kernel<<<wave_grid(nw), blk, 0, stream>>>(s16, CS, wqp_l, 144, 144, CS, 9,
        qp, nullptr, 144, 1.f, nw);
    gemm16_kernel<<<wave_grid(nw), blk, 0, stream>>>(s16, CS, wkp_l, 144, 144, CS, 9,
        kp, nullptr, 144, 1.f, nw);
    nw = MT * 18;
    gemm16_kernel<<<wave_grid(nw), blk, 0, stream>>>(s16, CS, wvp_l, 288, 288, CS, 18,
        vp, nullptr, 288, 1.f, nw);

    point_kernel<<<elem_grid(NT * H), blk, 0, stream>>>(rot, trans, qp, kp, vp,
        qpg, kpg16, vpg16, qn, kn);

    nw = NT * MT;
    bias_kernel<<<wave_grid(nw), blk, 0, stream>>>(z16, wb_l, biasb, nw);

    nw = H * MT * MT;
    score_kernel<<<wave_grid(nw), blk, 0, stream>>>(q16, k16, qpg, kpg16, qn, kn,
        biasb, hw_l, att, nw);

    softmax_kernel<<<dim3(H * NT), blk, 0, stream>>>(att, att16);

    nw = H * MT * 3;
    o_optg_kernel<<<wave_grid(nw), blk, 0, stream>>>(att16, v16, vpg16, obuf, optg, nw);

    nw = NT * (CZ / 16);
    opair_kernel<<<wave_grid(nw), blk, 0, stream>>>(att16, z16, opair, nw);

    cat_kernel<<<elem_grid((size_t)NT * CAT), blk, 0, stream>>>(obuf, optg, opair,
        rot, trans, cat16);

    nw = MT * (CS / 16);
    out_kernel<<<wave_grid(nw), blk, 0, stream>>>(cat16, wo_l, bo_l, s_out, s16, nw);
  }

  int nw = (NT * NT) / 16;
  bp_kernel<<<wave_grid(nw), blk, 0, stream>>>(z16, w1_16, bp_b1, bp_w2, bp_b2,
                                               bp_out, nw);
  torsion_kernel<<<wave_grid(MT), blk, 0, stream>>>(s16, tor16, tor_b, tor_out, MT);
}
